// EdgeMindLinear_86028194939447
// MI455X (gfx1250) — matrix operand order repaired and checked by reading, not compiled, not run on hardware
//
#include <hip/hip_runtime.h>

typedef __attribute__((ext_vector_type(16))) _Float16 v16h;
typedef __attribute__((ext_vector_type(8)))  _Float16 v8h;
typedef __attribute__((ext_vector_type(2)))  _Float16 v2h;
typedef __attribute__((ext_vector_type(8)))  float    v8f;
typedef __attribute__((ext_vector_type(4)))  float    vf4;
typedef __attribute__((ext_vector_type(4)))  int      vi4;

#define Mdim 8192
#define Kdim 4096
#define Ndim 11008
#define BM 128
#define BN 256
#define BK 32
#define PA  40   // A LDS row stride in halves (80B -> conflict-free)
#define PBB 48   // B LDS row stride in bytes  (48B -> conflict-free, 16B aligned)
#define NSTEP (Kdim / BK)

// fp32 pair -> f16 pair (single v_cvt_pk_f16_f32)
static __device__ __forceinline__ v2h cvt_pk(float a, float b) {
#if __has_builtin(__builtin_amdgcn_cvt_pkrtz)
  auto r = __builtin_amdgcn_cvt_pkrtz(a, b);
  v2h o; __builtin_memcpy(&o, &r, 4); return o;
#else
  v2h o; o[0] = (_Float16)a; o[1] = (_Float16)b; return o;
#endif
}

// 4 packed int8 -> 4 f16 via magic bias: half(0x6400 | (b^0x80)) == 1152 + b.
// Emits group order (b0,b2),(b1,b3); A staging applies the same (0,2,1,3)
// K-permutation, so the dot-product over K is unchanged.
static __device__ __forceinline__ void cvt_i8x4(unsigned w, v2h& p01, v2h& p23) {
  unsigned lo = (w & 0x00FF00FFu) ^ 0x64806480u;          // v_bitop3_b32
  unsigned hi = ((w >> 8) & 0x00FF00FFu) ^ 0x64806480u;   // lshr + v_bitop3_b32
  v2h a, b;
  __builtin_memcpy(&a, &lo, 4);
  __builtin_memcpy(&b, &hi, 4);
  const _Float16 c = (_Float16)(-1152.0f);
  const v2h bias = {c, c};
  p01 = a + bias;   // v_pk_add_f16
  p23 = b + bias;   // v_pk_add_f16
}

static __device__ __forceinline__ void async_b128_to_lds(unsigned lds_addr,
                                                         const void* gaddr) {
  // VDST = LDS byte address VGPR, VADDR = 64-bit global address (ISA §15.18 op 98)
  asm volatile("global_load_async_to_lds_b128 %0, %1, off"
               :: "v"(lds_addr), "v"(gaddr) : "memory");
}
static __device__ __forceinline__ void async_b128_to_lds_off16(unsigned lds_addr,
                                                               const void* gaddr) {
  // INST_OFFSET applies to both the LDS and global addresses (ISA §10.7 pseudocode)
  asm volatile("global_load_async_to_lds_b128 %0, %1, off offset:16"
               :: "v"(lds_addr), "v"(gaddr) : "memory");
}
static __device__ __forceinline__ void wait_async0() {
  asm volatile("s_wait_asynccnt 0x0" ::: "memory");
}

__global__ __launch_bounds__(256)
void q8gemm_wmma_f16(const float* __restrict__ X,
                     const signed char* __restrict__ Wq,
                     const float* __restrict__ Sc,
                     float* __restrict__ Out) {
  __shared__ __attribute__((aligned(16))) _Float16    Ab[2][BM * PA];
  __shared__ __attribute__((aligned(16))) signed char Bb[2][BN * PBB];

  const int t  = threadIdx.x;
  const int m0 = blockIdx.y * BM;
  const int n0 = blockIdx.x * BN;

  // A staging: thread t -> (row t>>1, 16-half chunk (t&1)*16)
  const int sr = t >> 1;
  const int sc = (t & 1) << 4;
  const float*       gA = X  + (size_t)(m0 + sr) * Kdim + sc;
  // B: thread t owns the 32-byte k-row of weight row n0+t (async DMA'd to LDS)
  const signed char* gB = Wq + (size_t)(n0 + t) * Kdim;

  // wave / fragment coordinates: 8 waves as 2(M) x 4(N), wave tile 64x64
  const int lane = t & 31;
  const int lr   = lane & 15;
  const int lh   = lane >> 4;
  const int wid  = t >> 5;
  const int wm   = (wid & 1) * 64;
  const int wn   = (wid >> 1) * 64;

  vf4 fA0, fA1, fA2, fA3;

  auto issue_async_B = [&](int ks, int buf) {
    const signed char* src = gB + (size_t)ks * BK;
    const unsigned l0 = (unsigned)(uintptr_t)&Bb[buf][t * PBB];
    async_b128_to_lds(l0, src);
    async_b128_to_lds_off16(l0, src);
  };

  auto stage_loadA = [&](int ks) {
    const float* ga = gA + (size_t)ks * BK;
    fA0 = *(const vf4*)(ga + 0);
    fA1 = *(const vf4*)(ga + 4);
    fA2 = *(const vf4*)(ga + 8);
    fA3 = *(const vf4*)(ga + 12);
  };

  auto stage_storeA = [&](int buf) {
    // 16 fp32 -> 16 f16, K-permuted (0,2,1,3) per 4-group (matches B cvt order)
    union { v8h v; v2h p[4]; } a0, a1;
    a0.p[0] = cvt_pk(fA0.x, fA0.z);  a0.p[1] = cvt_pk(fA0.y, fA0.w);
    a0.p[2] = cvt_pk(fA1.x, fA1.z);  a0.p[3] = cvt_pk(fA1.y, fA1.w);
    a1.p[0] = cvt_pk(fA2.x, fA2.z);  a1.p[1] = cvt_pk(fA2.y, fA2.w);
    a1.p[2] = cvt_pk(fA3.x, fA3.z);  a1.p[3] = cvt_pk(fA3.y, fA3.w);
    *(v8h*)&Ab[buf][sr * PA + sc]     = a0.v;
    *(v8h*)&Ab[buf][sr * PA + sc + 8] = a1.v;
  };

  v8f acc[4][4];
#pragma unroll
  for (int i = 0; i < 4; ++i)
#pragma unroll
    for (int j = 0; j < 4; ++j)
#pragma unroll
      for (int v = 0; v < 8; ++v) acc[i][j][v] = 0.0f;

  // ---- prologue: tile 0 ----
  issue_async_B(0, 0);
  stage_loadA(0);
  stage_storeA(0);
  wait_async0();
  __syncthreads();

#pragma unroll 1
  for (int ks = 0; ks < NSTEP; ++ks) {
    const int cur = ks & 1;

    if (ks + 1 < NSTEP) {
      issue_async_B(ks + 1, cur ^ 1);     // DMA next B tile straight into LDS
      stage_loadA(ks + 1);
      if (ks + 2 < NSTEP)
        __builtin_prefetch(gA + (size_t)(ks + 2) * BK, 0, 0);  // global_prefetch_b8
    }

    // ---- compute: 16 x v_wmma_f32_16x16x32_f16 per k-step per wave ----
    v16h afr[4];
#pragma unroll
    for (int i = 0; i < 4; ++i) {
      const _Float16* ap = &Ab[cur][(wm + i * 16 + lr) * PA + lh * 16];
      v8h lo = *(const v8h*)ap;
      v8h hi = *(const v8h*)(ap + 8);
      afr[i] = __builtin_shufflevector(lo, hi, 0,1,2,3,4,5,6,7,8,9,10,11,12,13,14,15);
    }
#pragma unroll
    for (int j = 0; j < 4; ++j) {
      // raw int8 fragment: one ds_load_b128, then packed magic conversion
      const signed char* bp = &Bb[cur][(wn + j * 16 + lr) * PBB + lh * 16];
      union { vi4 v; unsigned u[4]; } w;
      w.v = *(const vi4*)bp;
      union { v16h v; v2h p[8]; } bfr;
#pragma unroll
      for (int g = 0; g < 4; ++g)
        cvt_i8x4(w.u[g], bfr.p[2 * g], bfr.p[2 * g + 1]);
#pragma unroll
      for (int i = 0; i < 4; ++i)
        acc[i][j] = __builtin_amdgcn_wmma_f32_16x16x32_f16(
            false, afr[i], false, bfr.v, (short)0, acc[i][j], false, false);
    }

    if (ks + 1 < NSTEP) {
      stage_storeA(cur ^ 1);
      wait_async0();                      // B tile ks+1 resident before publish
    }
    __syncthreads();
  }

  // ---- epilogue: fuse per-channel dequant scale into the store ----
#pragma unroll
  for (int j = 0; j < 4; ++j) {
    const int nn = n0 + wn + j * 16 + lr;
    const float s = Sc[nn];
#pragma unroll
    for (int i = 0; i < 4; ++i) {
      const int mm = m0 + wm + i * 16 + lh * 8;
      float* op = Out + (size_t)mm * Ndim + nn;
#pragma unroll
      for (int v = 0; v < 8; ++v)
        op[(size_t)v * Ndim] = acc[i][j][v] * s;
    }
  }
}

extern "C" void kernel_launch(void* const* d_in, const int* in_sizes, int n_in,
                              void* d_out, int out_size, void* d_ws, size_t ws_size,
                              hipStream_t stream) {
  const float*       x   = (const float*)d_in[0];
  const signed char* wq  = (const signed char*)d_in[1];
  const float*       sc  = (const float*)d_in[2];
  float*             out = (float*)d_out;

  dim3 grid(Ndim / BN, Mdim / BM);  // 43 x 64 workgroups
  q8gemm_wmma_f16<<<grid, dim3(256), 0, stream>>>(x, wq, sc, out);
}
